// MDT_28724741276422
// MI455X (gfx1250) — matrix operand order repaired and checked by reading, not compiled, not run on hardware
//
#include <hip/hip_runtime.h>
#include <hip/hip_bf16.h>
#include <math.h>

// ---------------- model constants ----------------
#define BB     8
#define DD     1152
#define NTOK   256
#define LKEEP  179
#define NHEAD  16
#define HDIM   72
#define MLPH   4608
#define REL_ROWS 964
#define ATTN_SCALE 0.11785113019775793f   // 72^-0.5

typedef __attribute__((ext_vector_type(16))) _Float16 v16h;
typedef __attribute__((ext_vector_type(8)))  _Float16 v8h;
typedef __attribute__((ext_vector_type(8)))  float    v8f;

static inline int ceil_div(int a, int b) { return (a + b - 1) / b; }

// ---------------- activations ----------------
template<int ACT>
__device__ __forceinline__ float act_fn(float x) {
  if (ACT == 1) { // tanh-approx GELU
    float x3 = x * x * x;
    return 0.5f * x * (1.0f + tanhf(0.7978845608028654f * (x + 0.044715f * x3)));
  }
  if (ACT == 2) { // SiLU
    return x / (1.0f + expf(-x));
  }
  return x;
}

// ---------------- WMMA GEMM: C[M,N] = act(A[M,K] @ W[K,N] + bias) ----------------
// f32 streamed from global, converted to f16 in LDS, f32 accumulation.
// Block tile 64(M) x 128(N), 8 waves as 2x4, each wave owns a 32x32 tile
// (2x2 WMMA accumulators -> 4 v_wmma per K-step, A/B fragments reused 2x).
#define GEMM_BM 64
#define GEMM_BN 128
#define GEMM_BK 32
#define LDS_STRIDE 40   // 32 + 8 halfs pad, keeps 16B alignment per row

template<int ACT>
__global__ __launch_bounds__(256)
void gemm_wmma(const float* __restrict__ A, const float* __restrict__ W,
               const float* __restrict__ bias, float* __restrict__ C,
               int M, int N, int K)
{
  __shared__ __align__(16) _Float16 As[GEMM_BM * LDS_STRIDE];   // [m][k]
  __shared__ __align__(16) _Float16 Bs[GEMM_BN * LDS_STRIDE];   // transposed: [n][k]

  const int tid  = threadIdx.x;
  const int lane = tid & 31;
  const int wave = tid >> 5;
  const int wm   = wave >> 2;     // 0..1 -> 32-row half
  const int wn   = wave & 3;      // 0..3 -> 32-col quarter
  const int hi   = lane >> 4;     // 0..1
  const int ln16 = lane & 15;

  const int m0 = blockIdx.y * GEMM_BM;
  const int n0 = blockIdx.x * GEMM_BN;

  v8f acc00 = {}, acc01 = {}, acc10 = {}, acc11 = {};   // [mSub][nSub]

  // cooperative-load mapping (256 threads)
  const int arow = tid >> 2;              // 0..63   (8 A elems each)
  const int acol = (tid & 3) * 8;         // 0,8,16,24
  const int brow = tid >> 3;              // 0..31   (k in tile, 16 B elems each)
  const int bcol = (tid & 7) * 16;        // 0..112
  const int agr  = (m0 + arow < M) ? (m0 + arow) : (M - 1);
  const bool bvalid0 = (n0 + bcol)     < N;   // N is always a multiple of 8 here
  const bool bvalid1 = (n0 + bcol + 8) < N;

  for (int k0 = 0; k0 < K; k0 += GEMM_BK) {
    // ---- stage A tile (row-major) ----
    {
      const float4* p = (const float4*)(A + (size_t)agr * K + k0 + acol);
      float4 f0 = p[0], f1 = p[1];
      v8h h;
      h[0]=(_Float16)f0.x; h[1]=(_Float16)f0.y; h[2]=(_Float16)f0.z; h[3]=(_Float16)f0.w;
      h[4]=(_Float16)f1.x; h[5]=(_Float16)f1.y; h[6]=(_Float16)f1.z; h[7]=(_Float16)f1.w;
      *(v8h*)&As[arow * LDS_STRIDE + acol] = h;
    }
    // ---- stage B tile transposed ([n][k]) ----
    {
      const float* wrow = W + (size_t)(k0 + brow) * N + n0 + bcol;
      v8h h0 = {}, h1 = {};
      if (bvalid0) {
        const float4* p = (const float4*)wrow;
        float4 f0 = p[0], f1 = p[1];
        h0[0]=(_Float16)f0.x; h0[1]=(_Float16)f0.y; h0[2]=(_Float16)f0.z; h0[3]=(_Float16)f0.w;
        h0[4]=(_Float16)f1.x; h0[5]=(_Float16)f1.y; h0[6]=(_Float16)f1.z; h0[7]=(_Float16)f1.w;
      }
      if (bvalid1) {
        const float4* p = (const float4*)(wrow + 8);
        float4 f0 = p[0], f1 = p[1];
        h1[0]=(_Float16)f0.x; h1[1]=(_Float16)f0.y; h1[2]=(_Float16)f0.z; h1[3]=(_Float16)f0.w;
        h1[4]=(_Float16)f1.x; h1[5]=(_Float16)f1.y; h1[6]=(_Float16)f1.z; h1[7]=(_Float16)f1.w;
      }
      #pragma unroll
      for (int j = 0; j < 8; ++j) {
        Bs[(bcol + j)     * LDS_STRIDE + brow] = h0[j];
        Bs[(bcol + 8 + j) * LDS_STRIDE + brow] = h1[j];
      }
    }
    // prefetch next K-slab into near caches while current tiles are consumed
    if (k0 + GEMM_BK < K) {
      __builtin_prefetch(A + (size_t)agr * K + (k0 + GEMM_BK) + acol, 0, 3);
      if (bvalid0)
        __builtin_prefetch(W + (size_t)(k0 + GEMM_BK + brow) * N + (n0 + bcol), 0, 3);
    }
    __syncthreads();

    // ---- fragments per ISA layout ----
    // A 16x32: lane m = lane&15; e<8 -> K=hi*8+e ; e>=8 -> K=16+hi*8+(e-8)
    v16h afrag0, afrag1, bfrag0, bfrag1;
    {
      const _Float16* a0 = &As[(wm * 32 + ln16) * LDS_STRIDE + hi * 8];
      v8h lo = *(const v8h*)a0;
      v8h hh = *(const v8h*)(a0 + 16);
      #pragma unroll
      for (int e = 0; e < 8; ++e) { afrag0[e] = lo[e]; afrag0[e + 8] = hh[e]; }
      const _Float16* a1 = &As[(wm * 32 + 16 + ln16) * LDS_STRIDE + hi * 8];
      lo = *(const v8h*)a1;
      hh = *(const v8h*)(a1 + 16);
      #pragma unroll
      for (int e = 0; e < 8; ++e) { afrag1[e] = lo[e]; afrag1[e + 8] = hh[e]; }
      // B 32x16: lane n = lane&15; e<8 -> K=hi*8+e ; e>=8 -> K=16+hi*8+(e-8)
      const _Float16* b0 = &Bs[(wn * 32 + ln16) * LDS_STRIDE + hi * 8];
      v8h c0 = *(const v8h*)b0;
      v8h c1 = *(const v8h*)(b0 + 16);
      #pragma unroll
      for (int e = 0; e < 8; ++e) { bfrag0[e] = c0[e]; bfrag0[e + 8] = c1[e]; }
      const _Float16* b1 = &Bs[(wn * 32 + 16 + ln16) * LDS_STRIDE + hi * 8];
      c0 = *(const v8h*)b1;
      c1 = *(const v8h*)(b1 + 16);
      #pragma unroll
      for (int e = 0; e < 8; ++e) { bfrag1[e] = c0[e]; bfrag1[e + 8] = c1[e]; }
    }

    acc00 = __builtin_amdgcn_wmma_f32_16x16x32_f16(false, afrag0, false, bfrag0,
                                                   (short)0, acc00, false, false);
    acc10 = __builtin_amdgcn_wmma_f32_16x16x32_f16(false, afrag1, false, bfrag0,
                                                   (short)0, acc10, false, false);
    acc01 = __builtin_amdgcn_wmma_f32_16x16x32_f16(false, afrag0, false, bfrag1,
                                                   (short)0, acc01, false, false);
    acc11 = __builtin_amdgcn_wmma_f32_16x16x32_f16(false, afrag1, false, bfrag1,
                                                   (short)0, acc11, false, false);
    __syncthreads();
  }

  // ---- epilogue: C/D layout lane n = lane&15, m = hi*8 + v ----
  const int nA = n0 + wn * 32 + ln16;
  const int nB = nA + 16;
  const int mBase = m0 + wm * 32 + hi * 8;
  if (nA < N) {
    const float bv = bias ? bias[nA] : 0.0f;
    #pragma unroll
    for (int v = 0; v < 8; ++v) {
      int m = mBase + v;
      if (m < M)      C[(size_t)m * N + nA]        = act_fn<ACT>(acc00[v] + bv);
      if (m + 16 < M) C[(size_t)(m + 16) * N + nA] = act_fn<ACT>(acc10[v] + bv);
    }
  }
  if (nB < N) {
    const float bv = bias ? bias[nB] : 0.0f;
    #pragma unroll
    for (int v = 0; v < 8; ++v) {
      int m = mBase + v;
      if (m < M)      C[(size_t)m * N + nB]        = act_fn<ACT>(acc01[v] + bv);
      if (m + 16 < M) C[(size_t)(m + 16) * N + nB] = act_fn<ACT>(acc11[v] + bv);
    }
  }
}

static void gemm(const float* A, const float* W, const float* bias, float* C,
                 int M, int N, int K, int act, hipStream_t s)
{
  dim3 g(ceil_div(N, GEMM_BN), ceil_div(M, GEMM_BM));
  if (act == 1)      gemm_wmma<1><<<g, 256, 0, s>>>(A, W, bias, C, M, N, K);
  else if (act == 2) gemm_wmma<2><<<g, 256, 0, s>>>(A, W, bias, C, M, N, K);
  else               gemm_wmma<0><<<g, 256, 0, s>>>(A, W, bias, C, M, N, K);
}

// ---------------- 2D sin-cos positional embedding (on the fly) ----------------
__device__ __forceinline__ float pos_embed(int n, int d) {
  const int i = n >> 4, j = n & 15;   // n = i*16 + j ; first 576 dims use x(=j), rest use y(=i)
  float coord; int dd;
  if (d < 576) { coord = (float)j; dd = d; }
  else         { coord = (float)i; dd = d - 576; }
  const bool use_sin = dd < 288;
  if (!use_sin) dd -= 288;
  const float omega = powf(10000.0f, -(float)dd * (1.0f / 288.0f));
  const float a = coord * omega;
  return use_sin ? sinf(a) : cosf(a);
}

// ---------------- small kernels ----------------
__global__ void patch_embed_kernel(const float* __restrict__ x, const float* __restrict__ pw,
                                   const float* __restrict__ pb, float* __restrict__ tok)
{
  int idx = blockIdx.x * 256 + threadIdx.x;
  if (idx >= BB * NTOK * DD) return;
  const int d = idx % DD;
  const int n = (idx / DD) % NTOK;
  const int b = idx / (DD * NTOK);
  const int hh = n >> 4, ww = n & 15;
  float acc = pb[d];
  #pragma unroll
  for (int cpq = 0; cpq < 16; ++cpq) {
    const int c = cpq >> 2, p = (cpq >> 1) & 1, q = cpq & 1;
    const float xv = x[((b * 4 + c) * 32 + (hh * 2 + p)) * 32 + (ww * 2 + q)];
    acc += xv * pw[cpq * DD + d];
  }
  tok[idx] = acc + pos_embed(n, d);
}

__global__ void temb_kernel(const float* __restrict__ t, float* __restrict__ temb) {
  const int b = blockIdx.x, j = threadIdx.x;   // 128 threads
  const float f = expf(-logf(10000.0f) * (float)j / 128.0f);
  const float a = t[b] * f;
  temb[b * 256 + j]       = cosf(a);
  temb[b * 256 + 128 + j] = sinf(a);
}

__global__ void cond_kernel(const float* __restrict__ t2, const float* __restrict__ label_table,
                            const int* __restrict__ y, float* __restrict__ c, float* __restrict__ sc)
{
  int idx = blockIdx.x * 256 + threadIdx.x;
  if (idx >= BB * DD) return;
  const int b = idx / DD, j = idx % DD;
  const float v = t2[idx] + label_table[(size_t)y[b] * DD + j];
  c[idx] = v;
  sc[idx] = v / (1.0f + expf(-v));   // silu(c), reused by every block
}

__global__ __launch_bounds__(256)
void ln_mod_kernel(const float* __restrict__ x, const float* __restrict__ mod, int modStride,
                   int shiftOff, int scaleOff, float* __restrict__ yout, int L)
{
  __shared__ float red[256];
  const int row = blockIdx.x;          // 0 .. B*L-1
  const int b = row / L;
  const float* xr = x + (size_t)row * DD;
  float s = 0.f, s2 = 0.f;
  for (int j = threadIdx.x; j < DD; j += 256) { float v = xr[j]; s += v; s2 += v * v; }
  red[threadIdx.x] = s; __syncthreads();
  for (int o = 128; o > 0; o >>= 1) { if (threadIdx.x < o) red[threadIdx.x] += red[threadIdx.x + o]; __syncthreads(); }
  const float mean = red[0] * (1.0f / DD);
  __syncthreads();
  red[threadIdx.x] = s2; __syncthreads();
  for (int o = 128; o > 0; o >>= 1) { if (threadIdx.x < o) red[threadIdx.x] += red[threadIdx.x + o]; __syncthreads(); }
  const float var = red[0] * (1.0f / DD) - mean * mean;
  const float r = rsqrtf(var + 1e-6f);
  const float* mrow = mod + (size_t)b * modStride;
  for (int j = threadIdx.x; j < DD; j += 256) {
    const float v = (xr[j] - mean) * r;
    yout[(size_t)row * DD + j] = v * (1.0f + mrow[scaleOff + j]) + mrow[shiftOff + j];
  }
}

__global__ __launch_bounds__(256)
void attn_kernel(const float* __restrict__ qkv, const float* __restrict__ rel,
                 const int* __restrict__ ids_keep, float* __restrict__ out, int L)
{
  __shared__ float qv[HDIM];
  __shared__ float pr[256];
  __shared__ float red[256];
  const int tid = threadIdx.x;
  const int qi = blockIdx.x, h = blockIdx.y, b = blockIdx.z;
  if (tid < HDIM) qv[tid] = qkv[((size_t)b * L + qi) * (3 * DD) + h * HDIM + tid];
  __syncthreads();
  float s = -3.0e38f;
  if (tid < L) {
    const float* kr = qkv + ((size_t)b * L + tid) * (3 * DD) + DD + h * HDIM;
    float d = 0.f;
    #pragma unroll 8
    for (int e = 0; e < HDIM; ++e) d += qv[e] * kr[e];
    const int qp = ids_keep ? ids_keep[b * L + qi]  : qi;
    const int kp = ids_keep ? ids_keep[b * L + tid] : tid;
    const int di = (qp >> 4) - (kp >> 4);
    const int dj = (qp & 15) - (kp & 15);
    s = d * ATTN_SCALE + rel[((di + 15) * 31 + (dj + 15)) * NHEAD + h];
  }
  red[tid] = s; __syncthreads();
  for (int o = 128; o > 0; o >>= 1) { if (tid < o) red[tid] = fmaxf(red[tid], red[tid + o]); __syncthreads(); }
  const float mx = red[0];
  __syncthreads();
  const float p = (tid < L) ? expf(s - mx) : 0.f;
  pr[tid] = p;
  red[tid] = p; __syncthreads();
  for (int o = 128; o > 0; o >>= 1) { if (tid < o) red[tid] += red[tid + o]; __syncthreads(); }
  const float inv = 1.0f / red[0];
  if (tid < HDIM) {
    const float* vb = qkv + (size_t)b * L * (3 * DD) + 2 * DD + h * HDIM + tid;
    float acc = 0.f;
    for (int k = 0; k < L; ++k) acc += pr[k] * vb[(size_t)k * (3 * DD)];
    out[((size_t)b * L + qi) * DD + h * HDIM + tid] = acc * inv;
  }
}

__global__ void resid_gate_kernel(float* __restrict__ x, const float* __restrict__ mod,
                                  int modStride, int gOff, const float* __restrict__ h,
                                  int L, int total)
{
  int idx = blockIdx.x * 256 + threadIdx.x;
  if (idx >= total) return;
  const int j = idx % DD;
  const int b = idx / (DD * L);
  x[idx] += mod[(size_t)b * modStride + gOff + j] * h[idx];
}

__global__ void add_inplace_kernel(float* __restrict__ x, const float* __restrict__ a, int total) {
  int i = blockIdx.x * 256 + threadIdx.x;
  if (i < total) x[i] += a[i];
}

__global__ void gather_kernel(const float* __restrict__ tok, const int* __restrict__ ids_keep,
                              float* __restrict__ xm)
{
  int idx = blockIdx.x * 256 + threadIdx.x;
  if (idx >= BB * LKEEP * DD) return;
  const int j = idx % DD;
  const int l = (idx / DD) % LKEEP;
  const int b = idx / (DD * LKEEP);
  xm[idx] = tok[((size_t)b * NTOK + ids_keep[b * LKEEP + l]) * DD + j];
}

__global__ void restore_kernel(const float* __restrict__ xm, const int* __restrict__ ids_restore,
                               const float* __restrict__ mask_token,
                               float* __restrict__ xf, float* __restrict__ xbefore)
{
  int idx = blockIdx.x * 256 + threadIdx.x;
  if (idx >= BB * NTOK * DD) return;
  const int j = idx % DD;
  const int n = (idx / DD) % NTOK;
  const int b = idx / (DD * NTOK);
  const int idr = ids_restore[b * NTOK + n];
  float v = (idr < LKEEP) ? xm[((size_t)b * LKEEP + idr) * DD + j] : mask_token[j];
  v += pos_embed(n, j);
  xf[idx] = v;
  xbefore[idx] = v;
}

__global__ void maskmix_kernel(float* __restrict__ xf, const float* __restrict__ xbefore,
                               const int* __restrict__ ids_restore)
{
  int idx = blockIdx.x * 256 + threadIdx.x;
  if (idx >= BB * NTOK * DD) return;
  const int n = (idx / DD) % NTOK;
  const int b = idx / (DD * NTOK);
  if (ids_restore[b * NTOK + n] < LKEEP) xf[idx] = xbefore[idx];   // keep positions revert
}

__global__ void unpatch_kernel(const float* __restrict__ ftok, float* __restrict__ out) {
  int idx = blockIdx.x * 256 + threadIdx.x;
  if (idx >= BB * 8 * 32 * 32) return;
  const int j  = idx % 32;
  const int i  = (idx / 32) % 32;
  const int cc = (idx / 1024) % 8;
  const int b  = idx / (1024 * 8);
  const int hh = i >> 1, p = i & 1, ww = j >> 1, q = j & 1;
  out[idx] = ftok[((size_t)b * NTOK + hh * 16 + ww) * 32 + (p * 2 + q) * 8 + cc];
}

// ---------------- one transformer block ----------------
struct BlockBufs { float *mod, *xln, *qkv, *attn, *tmp, *h1; };

static void run_block(int i, float* x, int L, const int* ids_keep,
                      const float* qkv_w, const float* qkv_b,
                      const float* proj_w, const float* proj_b,
                      const float* fc1_w, const float* fc1_b,
                      const float* fc2_w, const float* fc2_b,
                      const float* ada_w, const float* ada_b,
                      const float* rel_table, const float* sc,
                      BlockBufs bf, hipStream_t s)
{
  const int M = BB * L;
  const int total = M * DD;
  // adaLN modulation: mod = silu(c) @ ada_w[i] + ada_b[i]   (B x 6912)
  gemm(sc, ada_w + (size_t)i * DD * 6912, ada_b + (size_t)i * 6912, bf.mod, BB, 6912, DD, 0, s);
  // attention branch
  ln_mod_kernel<<<M, 256, 0, s>>>(x, bf.mod, 6912, /*shift*/0, /*scale*/1152, bf.xln, L);
  gemm(bf.xln, qkv_w + (size_t)i * DD * 3 * DD, qkv_b + (size_t)i * 3 * DD, bf.qkv, M, 3 * DD, DD, 0, s);
  attn_kernel<<<dim3(L, NHEAD, BB), 256, 0, s>>>(bf.qkv, rel_table + (size_t)i * REL_ROWS * NHEAD,
                                                 ids_keep, bf.attn, L);
  gemm(bf.attn, proj_w + (size_t)i * DD * DD, proj_b + (size_t)i * DD, bf.tmp, M, DD, DD, 0, s);
  resid_gate_kernel<<<ceil_div(total, 256), 256, 0, s>>>(x, bf.mod, 6912, /*g1*/2304, bf.tmp, L, total);
  // MLP branch
  ln_mod_kernel<<<M, 256, 0, s>>>(x, bf.mod, 6912, /*shift*/3456, /*scale*/4608, bf.xln, L);
  gemm(bf.xln, fc1_w + (size_t)i * DD * MLPH, fc1_b + (size_t)i * MLPH, bf.h1, M, MLPH, DD, /*GELU*/1, s);
  gemm(bf.h1, fc2_w + (size_t)i * MLPH * DD, fc2_b + (size_t)i * DD, bf.tmp, M, DD, MLPH, 0, s);
  resid_gate_kernel<<<ceil_div(total, 256), 256, 0, s>>>(x, bf.mod, 6912, /*g2*/5760, bf.tmp, L, total);
}

// ---------------- entry ----------------
extern "C" void kernel_launch(void* const* d_in, const int* in_sizes, int n_in,
                              void* d_out, int out_size, void* d_ws, size_t ws_size,
                              hipStream_t stream)
{
  (void)in_sizes; (void)n_in; (void)out_size; (void)ws_size;
  // setup_inputs() dict order: x, t, params{...insertion order...}, y, ids_keep, ids_restore
  const float* x_in        = (const float*)d_in[0];
  const float* t_in        = (const float*)d_in[1];
  const float* patch_w     = (const float*)d_in[2];
  const float* patch_b     = (const float*)d_in[3];
  const float* t_w1        = (const float*)d_in[4];
  const float* t_b1        = (const float*)d_in[5];
  const float* t_w2        = (const float*)d_in[6];
  const float* t_b2        = (const float*)d_in[7];
  const float* label_table = (const float*)d_in[8];
  const float* qkv_w       = (const float*)d_in[9];
  const float* qkv_b       = (const float*)d_in[10];
  const float* proj_w      = (const float*)d_in[11];
  const float* proj_b      = (const float*)d_in[12];
  const float* fc1_w       = (const float*)d_in[13];
  const float* fc1_b       = (const float*)d_in[14];
  const float* fc2_w       = (const float*)d_in[15];
  const float* fc2_b       = (const float*)d_in[16];
  const float* ada_w       = (const float*)d_in[17];
  const float* ada_b       = (const float*)d_in[18];
  const float* rel_table   = (const float*)d_in[19];
  const float* mask_token  = (const float*)d_in[20];
  const float* fin_ada_w   = (const float*)d_in[21];
  const float* fin_ada_b   = (const float*)d_in[22];
  const float* fin_w       = (const float*)d_in[23];
  const float* fin_b       = (const float*)d_in[24];
  const int*   y_in        = (const int*)d_in[25];
  const int*   ids_keep    = (const int*)d_in[26];
  const int*   ids_restore = (const int*)d_in[27];

  // workspace bump allocator (all sizes multiples of 4 floats -> 16B alignment preserved)
  float* wsp = (float*)d_ws;
  auto alloc = [&](size_t nf) { float* p = wsp; wsp += nf; return p; };
  float* tok   = alloc((size_t)BB * NTOK * DD);
  float* xcur  = alloc((size_t)BB * NTOK * DD);
  float* xf    = alloc((size_t)BB * NTOK * DD);
  float* xbef  = alloc((size_t)BB * NTOK * DD);
  float* xln   = alloc((size_t)BB * NTOK * DD);
  float* qkvb  = alloc((size_t)BB * NTOK * 3 * DD);
  float* attnb = alloc((size_t)BB * NTOK * DD);
  float* tmpb  = alloc((size_t)BB * NTOK * DD);
  float* h1    = alloc((size_t)BB * NTOK * MLPH);
  float* skip0 = alloc((size_t)BB * LKEEP * DD);
  float* skip1 = alloc((size_t)BB * LKEEP * DD);
  float* skip2 = alloc((size_t)BB * LKEEP * DD);
  float* modb  = alloc((size_t)BB * 6912);
  float* temb  = alloc((size_t)BB * 256);
  float* th    = alloc((size_t)BB * DD);
  float* t2    = alloc((size_t)BB * DD);
  float* cbuf  = alloc((size_t)BB * DD);
  float* scb   = alloc((size_t)BB * DD);
  float* fmod  = alloc((size_t)BB * 2304);
  float* ftok  = alloc((size_t)BB * NTOK * 32);

  BlockBufs bf{modb, xln, qkvb, attnb, tmpb, h1};

  // 1) tokens = patch-embed + pos-embed
  patch_embed_kernel<<<ceil_div(BB * NTOK * DD, 256), 256, 0, stream>>>(x_in, patch_w, patch_b, tok);

  // 2) conditioning: timestep MLP + label table -> c, silu(c)
  temb_kernel<<<BB, 128, 0, stream>>>(t_in, temb);
  gemm(temb, t_w1, t_b1, th, BB, DD, 256, /*SiLU*/2, stream);
  gemm(th, t_w2, t_b2, t2, BB, DD, DD, 0, stream);
  cond_kernel<<<ceil_div(BB * DD, 256), 256, 0, stream>>>(t2, label_table, y_in, cbuf, scb);

  // 3) keep-token gather
  gather_kernel<<<ceil_div(BB * LKEEP * DD, 256), 256, 0, stream>>>(tok, ids_keep, xcur);

  const size_t skipBytes = (size_t)BB * LKEEP * DD * sizeof(float);
  const int totKeep = BB * LKEEP * DD;
  float* skips[3] = {skip0, skip1, skip2};

  // 4) encoder-in blocks 0..2 (push skip snapshots)
  for (int i = 0; i < 3; ++i) {
    run_block(i, xcur, LKEEP, ids_keep, qkv_w, qkv_b, proj_w, proj_b,
              fc1_w, fc1_b, fc2_w, fc2_b, ada_w, ada_b, rel_table, scb, bf, stream);
    hipMemcpyAsync(skips[i], xcur, skipBytes, hipMemcpyDeviceToDevice, stream);
  }
  // 5) encoder-out blocks 3..5 (pop skips in reverse)
  for (int i = 3; i < 6; ++i) {
    add_inplace_kernel<<<ceil_div(totKeep, 256), 256, 0, stream>>>(xcur, skips[5 - i], totKeep);
    run_block(i, xcur, LKEEP, ids_keep, qkv_w, qkv_b, proj_w, proj_b,
              fc1_w, fc1_b, fc2_w, fc2_b, ada_w, ada_b, rel_table, scb, bf, stream);
  }

  // 6) un-shuffle + mask tokens + pos-embed; keep a copy for the mask-mix
  const int totFull = BB * NTOK * DD;
  restore_kernel<<<ceil_div(totFull, 256), 256, 0, stream>>>(xcur, ids_restore, mask_token, xf, xbef);

  // 7) middle block (params index 8, full sequence, no mask)
  run_block(8, xf, NTOK, nullptr, qkv_w, qkv_b, proj_w, proj_b,
            fc1_w, fc1_b, fc2_w, fc2_b, ada_w, ada_b, rel_table, scb, bf, stream);
  maskmix_kernel<<<ceil_div(totFull, 256), 256, 0, stream>>>(xf, xbef, ids_restore);

  // 8) decoder blocks (params indices 6,7), each with full-token input skip
  for (int i = 6; i < 8; ++i) {
    add_inplace_kernel<<<ceil_div(totFull, 256), 256, 0, stream>>>(xf, tok, totFull);
    run_block(i, xf, NTOK, nullptr, qkv_w, qkv_b, proj_w, proj_b,
              fc1_w, fc1_b, fc2_w, fc2_b, ada_w, ada_b, rel_table, scb, bf, stream);
  }

  // 9) final adaLN + linear + unpatchify
  gemm(scb, fin_ada_w, fin_ada_b, fmod, BB, 2304, DD, 0, stream);
  ln_mod_kernel<<<BB * NTOK, 256, 0, stream>>>(xf, fmod, 2304, /*shift*/0, /*scale*/1152, xln, NTOK);
  gemm(xln, fin_w, fin_b, ftok, BB * NTOK, 32, DD, 0, stream);
  unpatch_kernel<<<ceil_div(BB * 8 * 32 * 32, 256), 256, 0, stream>>>(ftok, (float*)d_out);
}
